// VAE_42021960024327
// MI455X (gfx1250) — matrix operand order repaired and checked by reading, not compiled, not run on hardware
//
#include <hip/hip_runtime.h>
#include <hip/hip_bf16.h>

typedef __attribute__((ext_vector_type(16))) __bf16 v16bf;
typedef __attribute__((ext_vector_type(8)))  float  v8f;
typedef __attribute__((ext_vector_type(4)))  int    v4i;

__device__ __forceinline__ unsigned short f2bf(float f) {
  unsigned int u = __float_as_uint(f);
  u += 0x7FFFu + ((u >> 16) & 1u);
  return (unsigned short)(u >> 16);
}

union FragU { v16bf v; uint4 q[2]; };

#define PA 40   // LDS pitch in ushorts for 32-K tiles (+8 pad vs bank conflicts)
#define PB 40

#if __has_builtin(__builtin_amdgcn_global_load_async_to_lds_b128)
#define HAVE_ASYNC_LDS 1
#else
#define HAVE_ASYNC_LDS 0
#endif

// 16-byte global -> LDS copy: async (ASYNCcnt) on gfx1250 if available.
__device__ __forceinline__ void copy16(const unsigned short* g, unsigned short* l) {
#if HAVE_ASYNC_LDS
  unsigned short* gnc = const_cast<unsigned short*>(g);
  __builtin_amdgcn_global_load_async_to_lds_b128(
      (__attribute__((address_space(1))) v4i*)gnc,
      (__attribute__((address_space(3))) v4i*)l, 0, 0);
#else
  *reinterpret_cast<uint4*>(l) = *reinterpret_cast<const uint4*>(g);
#endif
}

__device__ __forceinline__ void wait_async_copies() {
#if HAVE_ASYNC_LDS
#if __has_builtin(__builtin_amdgcn_s_wait_asynccnt)
  __builtin_amdgcn_s_wait_asynccnt(0);
#else
  asm volatile("s_wait_asynccnt 0" ::: "memory");
#endif
#endif
}

// C = act( A[R,Ka](bf16) @ Bt[Cout,Ka]^T(bf16) + bias (+ residual fp32) )
// Ka is a multiple of 32 (zero-padded at pre-conversion -> no tail path).
// Block 128x128, 8 waves, wave tile 32x64 -> 2x4 v_wmma_f32_16x16x32_bf16.
__global__ __launch_bounds__(256)
void gemm_wmma_bf16(const unsigned short* __restrict__ A,
                    const unsigned short* __restrict__ Bt,
                    const float* __restrict__ bias,
                    const float* __restrict__ residual,
                    float* __restrict__ Cf, unsigned short* __restrict__ Cb,
                    int R, int Ka, int Cout, int relu)
{
  __shared__ unsigned short lA[128 * PA];
  __shared__ unsigned short lB[128 * PB];

  const int t  = threadIdx.x;
  const int c0 = blockIdx.x * 128;
  const int r0 = blockIdx.y * 128;

  const int lane = t & 31;
  const int wid  = t >> 5;
  const int wr = (wid >> 1) * 32;
  const int wc = (wid & 1) * 64;
  const int lm = lane & 15;
  const int kh = lane >> 4;

  v8f acc[2][4];
#pragma unroll
  for (int mt = 0; mt < 2; ++mt)
#pragma unroll
    for (int nt = 0; nt < 4; ++nt)
      acc[mt][nt] = (v8f){0.f,0.f,0.f,0.f,0.f,0.f,0.f,0.f};

  // staging: thread pair per 64B row (A) / col (Bt); each thread copies 32B
  const int srow  = t >> 1;
  const int shalf = (t & 1) * 16;
  const unsigned short* gA = A  + (size_t)(r0 + srow) * Ka + shalf;
  const unsigned short* gB = Bt + (size_t)(c0 + srow) * Ka + shalf;
  unsigned short* lAd = &lA[srow * PA + shalf];
  unsigned short* lBd = &lB[srow * PB + shalf];

  for (int k0 = 0; k0 < Ka; k0 += 32) {
    copy16(gA + k0,     lAd);
    copy16(gA + k0 + 8, lAd + 8);
    copy16(gB + k0,     lBd);
    copy16(gB + k0 + 8, lBd + 8);
    wait_async_copies();
    __syncthreads();

    FragU af[2], bfr[4];
#pragma unroll
    for (int mt = 0; mt < 2; ++mt) {
      const uint4* p = reinterpret_cast<const uint4*>(&lA[(wr + mt * 16 + lm) * PA]);
      af[mt].q[0] = p[kh];        // K = 8*kh + 0..7
      af[mt].q[1] = p[2 + kh];    // K = 16 + 8*kh + 0..7
    }
#pragma unroll
    for (int nt = 0; nt < 4; ++nt) {
      const uint4* p = reinterpret_cast<const uint4*>(&lB[(wc + nt * 16 + lm) * PB]);
      bfr[nt].q[0] = p[kh];           // K = 8*kh + 0..7
      bfr[nt].q[1] = p[2 + kh];       // K = 16 + 8*kh + 0..7
    }
#pragma unroll
    for (int mt = 0; mt < 2; ++mt)
#pragma unroll
      for (int nt = 0; nt < 4; ++nt)
        acc[mt][nt] = __builtin_amdgcn_wmma_f32_16x16x32_bf16(
            false, af[mt].v, false, bfr[nt].v, (short)0, acc[mt][nt], false, false);
    __syncthreads();
  }

#pragma unroll
  for (int mt = 0; mt < 2; ++mt) {
#pragma unroll
    for (int nt = 0; nt < 4; ++nt) {
      int col = c0 + wc + nt * 16 + lm;
      float bb = bias ? bias[col] : 0.0f;
      int rowBase = r0 + wr + mt * 16 + 8 * kh;
#pragma unroll
      for (int r = 0; r < 8; ++r) {
        int row = rowBase + r;
        float v = acc[mt][nt][r] + bb;
        if (residual) v += residual[(size_t)row * Cout + col];
        if (relu) v = fmaxf(v, 0.0f);
        if (Cf) Cf[(size_t)row * Cout + col] = v;
        if (Cb) Cb[(size_t)row * Cout + col] = f2bf(v);
      }
    }
  }
}

// One-time: Wt[c][k] = bf16(W[k][c]), zero-padded to Kp.
__global__ void wtrans_bf16(const float* __restrict__ W, unsigned short* __restrict__ Wt,
                            int K, int C, int Kp)
{
  int idx = blockIdx.x * blockDim.x + threadIdx.x;
  if (idx >= C * Kp) return;
  int c = idx / Kp, k = idx - c * Kp;
  Wt[idx] = (k < K) ? f2bf(W[(size_t)k * C + c]) : (unsigned short)0;
}

__global__ void tobf16(const float* __restrict__ s, unsigned short* __restrict__ d, int n)
{
  int i = blockIdx.x * blockDim.x + threadIdx.x;
  if (i < n) d[i] = f2bf(s[i]);
}

// Tiny-output heads (M = 4 or 1, K = 256): one thread per output element.
__global__ void head_gemm(const float* __restrict__ A, const float* __restrict__ W,
                          const float* __restrict__ bias, float* __restrict__ C,
                          int R, int K, int M)
{
  int idx = blockIdx.x * blockDim.x + threadIdx.x;
  if (idx >= R * M) return;
  int n = idx / M, m = idx - n * M;
  const float* a = A + (size_t)n * K;
  float s = bias[m];
  for (int k = 0; k < K; k += 4) {
    s = fmaf(a[k],     W[(size_t)k       * M + m], s);
    s = fmaf(a[k + 1], W[(size_t)(k + 1) * M + m], s);
    s = fmaf(a[k + 2], W[(size_t)(k + 2) * M + m], s);
    s = fmaf(a[k + 3], W[(size_t)(k + 3) * M + m], s);
  }
  C[idx] = s;
}

// dec_aux2 layer 0: [R,9] @ [9,M] + b, relu  -> bf16 activation
__global__ void dense9_relu(const float* __restrict__ A9, const float* __restrict__ W,
                            const float* __restrict__ bias, unsigned short* __restrict__ C,
                            int R, int M)
{
  int idx = blockIdx.x * blockDim.x + threadIdx.x;
  if (idx >= R * M) return;
  int n = idx / M, m = idx - n * M;
  const float* a = A9 + (size_t)n * 9;
  float s = bias[m];
#pragma unroll
  for (int k = 0; k < 9; ++k) s = fmaf(a[k], W[(size_t)k * M + m], s);
  C[idx] = f2bf(fmaxf(s, 0.0f));
}

// cat1b[n] = bf16([x[n] | z1_mean[n] | z2_mean[n] | 0-pad]) with pitch 1056
__global__ void concat_unmix_bf16(const float* __restrict__ x, const float* __restrict__ z1m,
                                  const float* __restrict__ z2m, unsigned short* __restrict__ out,
                                  int R)
{
  int idx = blockIdx.x * blockDim.x + threadIdx.x;
  if (idx >= R * 1056) return;
  int n = idx / 1056, c = idx - n * 1056;
  float v;
  if (c < 1024)      v = x[(size_t)n * 1024 + c];
  else if (c < 1028) v = z1m[n * 4 + (c - 1024)];
  else if (c < 1032) v = z2m[n * 4 + (c - 1028)];
  else               v = 0.0f;
  out[idx] = f2bf(v);
}

// Reparameterization + softplus + build cat9 = [omega | z1 | z2]
__global__ void sampler(const float* __restrict__ omm, const float* __restrict__ omv,
                        const float* __restrict__ z1m, const float* __restrict__ z1v,
                        const float* __restrict__ z2m, const float* __restrict__ z2v,
                        const float* __restrict__ eo,  const float* __restrict__ e1,
                        const float* __restrict__ e2,
                        float* __restrict__ omega, float* __restrict__ osq,
                        float* __restrict__ z1s, float* __restrict__ z2s,
                        float* __restrict__ cat9, int R)
{
  int n = blockIdx.x * blockDim.x + threadIdx.x;
  if (n >= R) return;
  float m  = omm[n];
  float sp = (m > 20.0f) ? m : log1pf(__expf(m));
  float om = sp + __expf(0.5f * omv[n]) * eo[n];
  omega[n] = om;
  osq[n]   = om * om;
  cat9[n * 9 + 0] = om;
#pragma unroll
  for (int d = 0; d < 4; ++d) {
    float a = z1m[n * 4 + d] + __expf(0.5f * z1v[n * 4 + d]) * e1[n * 4 + d];
    z1s[n * 4 + d] = a;
    cat9[n * 9 + 1 + d] = a;
    float b = z2m[n * 4 + d] + __expf(0.5f * z2v[n * 4 + d]) * e2[n * 4 + d];
    z2s[n * 4 + d] = b;
    cat9[n * 9 + 5 + d] = b;
  }
}

__device__ __forceinline__ void odef(float a0, float a1, float tt, float osq,
                                     const float* __restrict__ bp,
                                     const float* __restrict__ wy0,
                                     const float* __restrict__ wy1,
                                     const float* __restrict__ wt,
                                     const float* __restrict__ w2,
                                     float b1s, float& k0, float& k1)
{
  float acc = b1s;
#pragma unroll 8
  for (int j = 0; j < 64; ++j) {
    float hh = fmaf(tt, wt[j], fmaf(a1, wy1[j], fmaf(a0, wy0[j], bp[j])));
    acc = fmaf(fmaxf(hh, 0.0f), w2[j], acc);
  }
  k0 = a1;
  k1 = acc - osq * __sinf(a0);
}

// RK4 scan: 1023 steps, one thread per sample; z1-part of dec_aux1 layer0 is
// time-invariant -> per-sample "base" in LDS (pitch 65 avoids bank conflicts).
__global__ __launch_bounds__(128)
void ode_rk4(const float* __restrict__ x, const float* __restrict__ osqA,
             const float* __restrict__ z1, const float* __restrict__ W0,
             const float* __restrict__ b0, const float* __restrict__ W1,
             const float* __restrict__ b1, float* __restrict__ yseq, int R)
{
  __shared__ float wy0[64], wy1[64], wt[64], w2[64];
  __shared__ float base[128 * 65];
  int t = threadIdx.x;
  if (t < 64) { wy0[t] = W0[256 + t]; wy1[t] = W0[320 + t]; wt[t] = W0[384 + t]; w2[t] = W1[t]; }
  int n = blockIdx.x * 128 + t;
  if (n < R) {
    float za = z1[n * 4 + 0], zb = z1[n * 4 + 1], zc = z1[n * 4 + 2], zd = z1[n * 4 + 3];
    for (int j = 0; j < 64; ++j)
      base[t * 65 + j] = fmaf(zd, W0[192 + j],
                         fmaf(zc, W0[128 + j],
                         fmaf(zb, W0[64 + j],
                         fmaf(za, W0[j], b0[j]))));
  }
  __syncthreads();
  if (n >= R) return;
  const float* bp = &base[t * 65];
  float osq = osqA[n];
  float b1s = b1[0];
  float y0 = x[(size_t)n * 1024], y1 = 0.0f;
  float* yp = yseq + (size_t)n * 1024;
  yp[0] = y0;
  const float h = 0.01f, h2 = 0.005f;
  for (int i = 0; i < 1023; ++i) {
    float tt = 0.01f * (float)i;
    float k1a, k1b, k2a, k2b, k3a, k3b, k4a, k4b;
    odef(y0,            y1,            tt,      osq, bp, wy0, wy1, wt, w2, b1s, k1a, k1b);
    odef(y0 + h2 * k1a, y1 + h2 * k1b, tt + h2, osq, bp, wy0, wy1, wt, w2, b1s, k2a, k2b);
    odef(y0 + h2 * k2a, y1 + h2 * k2b, tt + h2, osq, bp, wy0, wy1, wt, w2, b1s, k3a, k3b);
    odef(y0 + h  * k3a, y1 + h  * k3b, tt + h,  osq, bp, wy0, wy1, wt, w2, b1s, k4a, k4b);
    y0 += (h / 6.0f) * (k1a + 2.0f * k2a + 2.0f * k3a + k4a);
    y1 += (h / 6.0f) * (k1b + 2.0f * k2b + 2.0f * k3b + k4b);
    yp[i + 1] = y0;
  }
}

extern "C" void kernel_launch(void* const* d_in, const int* in_sizes, int n_in,
                              void* d_out, int out_size, void* d_ws, size_t ws_size,
                              hipStream_t stream)
{
  (void)in_sizes; (void)n_in; (void)out_size; (void)ws_size;
  const float* x  = (const float*)d_in[0];
  const float* eo = (const float*)d_in[1];
  const float* e1 = (const float*)d_in[2];
  const float* e2 = (const float*)d_in[3];
  const float* fa1W0 = (const float*)d_in[4];  const float* fa1b0 = (const float*)d_in[5];
  const float* fa1W1 = (const float*)d_in[6];  const float* fa1b1 = (const float*)d_in[7];
  const float* z1mW0 = (const float*)d_in[8];  const float* z1mb0 = (const float*)d_in[9];
  const float* z1mW1 = (const float*)d_in[10]; const float* z1mb1 = (const float*)d_in[11];
  const float* z1vW0 = (const float*)d_in[12]; const float* z1vb0 = (const float*)d_in[13];
  const float* z1vW1 = (const float*)d_in[14]; const float* z1vb1 = (const float*)d_in[15];
  const float* fa2W0 = (const float*)d_in[16]; const float* fa2b0 = (const float*)d_in[17];
  const float* fa2W1 = (const float*)d_in[18]; const float* fa2b1 = (const float*)d_in[19];
  const float* z2mW0 = (const float*)d_in[20]; const float* z2mb0 = (const float*)d_in[21];
  const float* z2mW1 = (const float*)d_in[22]; const float* z2mb1 = (const float*)d_in[23];
  const float* z2vW0 = (const float*)d_in[24]; const float* z2vb0 = (const float*)d_in[25];
  const float* z2vW1 = (const float*)d_in[26]; const float* z2vb1 = (const float*)d_in[27];
  const float* unW0  = (const float*)d_in[28]; const float* unb0  = (const float*)d_in[29];
  const float* unW1  = (const float*)d_in[30]; const float* unb1  = (const float*)d_in[31];
  const float* fpW0  = (const float*)d_in[32]; const float* fpb0  = (const float*)d_in[33];
  const float* fpW1  = (const float*)d_in[34]; const float* fpb1  = (const float*)d_in[35];
  const float* omW0  = (const float*)d_in[36]; const float* omb0  = (const float*)d_in[37];
  const float* omW1  = (const float*)d_in[38]; const float* omb1  = (const float*)d_in[39];
  const float* ovW0  = (const float*)d_in[40]; const float* ovb0  = (const float*)d_in[41];
  const float* ovW1  = (const float*)d_in[42]; const float* ovb1  = (const float*)d_in[43];
  const float* daW0  = (const float*)d_in[44]; const float* dab0  = (const float*)d_in[45];
  const float* daW1  = (const float*)d_in[46]; const float* dab1  = (const float*)d_in[47];
  const float* d2W0  = (const float*)d_in[48]; const float* d2b0  = (const float*)d_in[49];
  const float* d2W1  = (const float*)d_in[50]; const float* d2b1  = (const float*)d_in[51];

  const int R = 2048;
  char* w = (char*)d_ws;
  auto allocF = [&](size_t n) { float* p = (float*)w; w += n * sizeof(float); return p; };
  auto allocH = [&](size_t n) { unsigned short* p = (unsigned short*)w; w += n * sizeof(unsigned short); return p; };

  // fp32 scratch
  float* h256 = allocF((size_t)R * 256);
  float* yseq = allocF((size_t)R * 1024);
  float* z1m  = allocF(R * 4); float* z1v = allocF(R * 4);
  float* z2m  = allocF(R * 4); float* z2v = allocF(R * 4);
  float* z1s  = allocF(R * 4); float* z2s = allocF(R * 4);
  float* omm  = allocF(R);     float* omv = allocF(R);
  float* omeg = allocF(R);     float* osq = allocF(R);
  float* cat9 = allocF(R * 12);
  // bf16 activations
  unsigned short* xb    = allocH((size_t)R * 1024);
  unsigned short* bufAb = allocH((size_t)R * 1024);
  unsigned short* bufBb = allocH((size_t)R * 1024);
  unsigned short* cat1b = allocH((size_t)R * 1056);
  // bf16 transposed weights [Cout][Kp]
  unsigned short* fa1W0t = allocH(1024 * 1024);
  unsigned short* fa1W1t = allocH(1024 * 1024);
  unsigned short* fa2W0t = allocH(1024 * 1024);
  unsigned short* fa2W1t = allocH(1024 * 1024);
  unsigned short* unW0t  = allocH(1024 * 1056);
  unsigned short* unW1t  = allocH(1024 * 1024);
  unsigned short* fpW0t  = allocH(1024 * 1024);
  unsigned short* fpW1t  = allocH(1024 * 1024);
  unsigned short* d2W1t  = allocH(1024 * 1024);
  unsigned short* z1mW0t = allocH(256 * 1024);
  unsigned short* z1vW0t = allocH(256 * 1024);
  unsigned short* z2mW0t = allocH(256 * 1024);
  unsigned short* z2vW0t = allocH(256 * 1024);
  unsigned short* omW0t  = allocH(256 * 1024);
  unsigned short* ovW0t  = allocH(256 * 1024);

  auto wt = [&](const float* W, unsigned short* Wt, int K, int C, int Kp) {
    wtrans_bf16<<<(C * Kp + 255) / 256, 256, 0, stream>>>(W, Wt, K, C, Kp);
  };
  wt(fa1W0, fa1W0t, 1024, 1024, 1024);
  wt(fa1W1, fa1W1t, 1024, 1024, 1024);
  wt(fa2W0, fa2W0t, 1024, 1024, 1024);
  wt(fa2W1, fa2W1t, 1024, 1024, 1024);
  wt(unW0,  unW0t,  1032, 1024, 1056);
  wt(unW1,  unW1t,  1024, 1024, 1024);
  wt(fpW0,  fpW0t,  1024, 1024, 1024);
  wt(fpW1,  fpW1t,  1024, 1024, 1024);
  wt(d2W1,  d2W1t,  1024, 1024, 1024);
  wt(z1mW0, z1mW0t, 1024, 256, 1024);
  wt(z1vW0, z1vW0t, 1024, 256, 1024);
  wt(z2mW0, z2mW0t, 1024, 256, 1024);
  wt(z2vW0, z2vW0t, 1024, 256, 1024);
  wt(omW0,  omW0t,  1024, 256, 1024);
  wt(ovW0,  ovW0t,  1024, 256, 1024);
  tobf16<<<(R * 1024 + 255) / 256, 256, 0, stream>>>(x, xb, R * 1024);

  auto gemm = [&](const unsigned short* A, const unsigned short* Bt, const float* b,
                  const float* res, float* Cf, unsigned short* Cb, int Ka, int Cout, int relu) {
    gemm_wmma_bf16<<<dim3(Cout / 128, R / 128), 256, 0, stream>>>(
        A, Bt, b, res, Cf, Cb, R, Ka, Cout, relu);
  };

  // feat_aux1 -> z1 heads
  gemm(xb,    fa1W0t, fa1b0, nullptr, nullptr, bufAb, 1024, 1024, 1);
  gemm(bufAb, fa1W1t, fa1b1, nullptr, nullptr, bufBb, 1024, 1024, 1);
  gemm(bufBb, z1mW0t, z1mb0, nullptr, h256, nullptr, 1024, 256, 1);
  head_gemm<<<(R * 4 + 255) / 256, 256, 0, stream>>>(h256, z1mW1, z1mb1, z1m, R, 256, 4);
  gemm(bufBb, z1vW0t, z1vb0, nullptr, h256, nullptr, 1024, 256, 1);
  head_gemm<<<(R * 4 + 255) / 256, 256, 0, stream>>>(h256, z1vW1, z1vb1, z1v, R, 256, 4);

  // feat_aux2 -> z2 heads
  gemm(xb,    fa2W0t, fa2b0, nullptr, nullptr, bufAb, 1024, 1024, 1);
  gemm(bufAb, fa2W1t, fa2b1, nullptr, nullptr, bufBb, 1024, 1024, 1);
  gemm(bufBb, z2mW0t, z2mb0, nullptr, h256, nullptr, 1024, 256, 1);
  head_gemm<<<(R * 4 + 255) / 256, 256, 0, stream>>>(h256, z2mW1, z2mb1, z2m, R, 256, 4);
  gemm(bufBb, z2vW0t, z2vb0, nullptr, h256, nullptr, 1024, 256, 1);
  head_gemm<<<(R * 4 + 255) / 256, 256, 0, stream>>>(h256, z2vW1, z2vb1, z2v, R, 256, 4);

  // unmixer (Ka = 1056, zero-padded) + residual x
  concat_unmix_bf16<<<(R * 1056 + 255) / 256, 256, 0, stream>>>(x, z1m, z2m, cat1b, R);
  gemm(cat1b, unW0t, unb0, nullptr, nullptr, bufAb, 1056, 1024, 1);
  gemm(bufAb, unW1t, unb1, x,       nullptr, bufBb, 1024, 1024, 0);   // bufBb = unmixed

  // feat_phy -> omega heads
  gemm(bufBb, fpW0t, fpb0, nullptr, nullptr, bufAb, 1024, 1024, 1);
  gemm(bufAb, fpW1t, fpb1, nullptr, nullptr, bufBb, 1024, 1024, 1);   // bufBb = fp
  gemm(bufBb, omW0t, omb0, nullptr, h256, nullptr, 1024, 256, 1);
  head_gemm<<<(R + 255) / 256, 256, 0, stream>>>(h256, omW1, omb1, omm, R, 256, 1);
  gemm(bufBb, ovW0t, ovb0, nullptr, h256, nullptr, 1024, 256, 1);
  head_gemm<<<(R + 255) / 256, 256, 0, stream>>>(h256, ovW1, ovb1, omv, R, 256, 1);

  // reparameterize + RK4 scan
  sampler<<<(R + 255) / 256, 256, 0, stream>>>(omm, omv, z1m, z1v, z2m, z2v,
                                               eo, e1, e2, omeg, osq, z1s, z2s, cat9, R);
  ode_rk4<<<R / 128, 128, 0, stream>>>(x, osq, z1s, daW0, dab0, daW1, dab1, yseq, R);

  // dec_aux2 + residual y_seq -> output
  dense9_relu<<<(R * 1024 + 255) / 256, 256, 0, stream>>>(cat9, d2W0, d2b0, bufAb, R, 1024);
  gemm(bufAb, d2W1t, d2b1, yseq, (float*)d_out, nullptr, 1024, 1024, 0);
}
